// PointNet2Multiview2Plus_7954279432403
// MI455X (gfx1250) — matrix operand order repaired and checked by reading, not compiled, not run on hardware
//
// PointNet++ multiview forward for MI455X (gfx1250), compile-only target.
// MLP GEMMs run on v_wmma_f32_16x16x32_f16 with guard-free, vectorized f16
// fragment loads (K padded to 32 by construction, activations kept as f32 +
// f16 side copies). GEMM is templated on the N-subtile count so the inner
// loop is a branch-free clause of global_load_b128 -> v_wmma.

#include <hip/hip_runtime.h>

typedef _Float16 v16h __attribute__((ext_vector_type(16)));
typedef _Float16 v8h  __attribute__((ext_vector_type(8)));
typedef float    v8f  __attribute__((ext_vector_type(8)));

constexpr int Bc   = 2;        // batch
constexpr int NPts = 8192;     // points
constexpr int Vv   = 3;        // views
constexpr int Cimg = 128;      // image channels
constexpr int HWi  = 32 * 41;  // H*W
constexpr int NCls = 21;

static inline int padK(int k) { return (k + 31) & ~31; }

// ---------------------------------------------------------------- transpose (B,C,N)->(B,N,C)
__global__ void k_transpose(const float* __restrict__ in, float* __restrict__ out,
                            int C, int Np) {
  int t = blockIdx.x * blockDim.x + threadIdx.x;
  if (t >= Bc * C * Np) return;
  int n = t % Np; int bc = t / Np; int c = bc % C; int b = bc / C;
  out[((size_t)b * Np + n) * C + c] = in[t];
}

// ---------------------------------------------------------------- project_views
__global__ void k_fill_i32(int* __restrict__ p, int v, int n) {
  int t = blockIdx.x * blockDim.x + threadIdx.x;
  if (t < n) p[t] = v;
}

// last-write-wins scatter: keep max source index j per (b,v,point)
__global__ void k_scatter_last(const int* __restrict__ idx3d, int* __restrict__ jmax) {
  int t = blockIdx.x * blockDim.x + threadIdx.x;
  if (t >= Bc * Vv * NPts) return;
  int bv = t / NPts; int j = t % NPts;
  atomicMax(&jmax[bv * NPts + idx3d[t]], j);
}

// imgf[b,n,c] = max_v (assigned ? imageft[b,v,c,idx2d[j]] : 0)
__global__ void k_project(const float* __restrict__ imageft, const int* __restrict__ idx2d,
                          const int* __restrict__ jmax, float* __restrict__ imgf) {
  int t = blockIdx.x * blockDim.x + threadIdx.x;
  if (t >= Bc * NPts * Cimg) return;
  int c = t % Cimg; int bn = t / Cimg; int b = bn / NPts; int n = bn % NPts;
  float m = -3.4e38f;
  for (int v = 0; v < Vv; ++v) {
    int bv = b * Vv + v;
    int j = jmax[bv * NPts + n];
    float val = 0.f;
    if (j >= 0) val = imageft[((size_t)bv * Cimg + c) * HWi + idx2d[bv * NPts + j]];
    m = fmaxf(m, val);
  }
  imgf[t] = m;
}

// ---------------------------------------------------------------- farthest point sampling
// One workgroup per batch; dist[] lives in LDS (Npts <= 8192 -> 32KB of 320KB WGP LDS).
__global__ __launch_bounds__(256) void k_fps(const float* __restrict__ xyz,
                                             int* __restrict__ fidx, int Np, int S) {
  int b = blockIdx.x;
  const float* P = xyz + (size_t)b * Np * 3;
  __shared__ float dist[8192];
  __shared__ float rv[256];
  __shared__ int   ri[256];
  __shared__ int   s_far;
  for (int i = threadIdx.x; i < Np; i += blockDim.x) dist[i] = 1e10f;
  if (threadIdx.x == 0) s_far = 0;
  __syncthreads();
  for (int s = 0; s < S; ++s) {
    int far = s_far;
    if (threadIdx.x == 0) fidx[b * S + s] = far;   // scan outputs far BEFORE update
    float cx = P[far * 3 + 0], cy = P[far * 3 + 1], cz = P[far * 3 + 2];
    float best = -1.f; int bi = 0;
    for (int i = threadIdx.x; i < Np; i += blockDim.x) {
      float dx = P[i * 3 + 0] - cx, dy = P[i * 3 + 1] - cy, dz = P[i * 3 + 2] - cz;
      float d  = dx * dx + dy * dy + dz * dz;
      float dm = fminf(dist[i], d);
      dist[i]  = dm;
      if (dm > best) { best = dm; bi = i; }
    }
    rv[threadIdx.x] = best; ri[threadIdx.x] = bi;
    __syncthreads();
    for (int off = 128; off > 0; off >>= 1) {
      if ((int)threadIdx.x < off) {
        float v2 = rv[threadIdx.x + off]; int i2 = ri[threadIdx.x + off];
        float v1 = rv[threadIdx.x];       int i1 = ri[threadIdx.x];
        if (v2 > v1 || (v2 == v1 && i2 < i1)) { rv[threadIdx.x] = v2; ri[threadIdx.x] = i2; }
      }
      __syncthreads();
    }
    if (threadIdx.x == 0) s_far = ri[0];
    __syncthreads();
  }
}

// ---------------------------------------------------------------- ball query (ordered scan == sort trick)
__global__ void k_query_ball(const float* __restrict__ xyz, const int* __restrict__ fidx,
                             float r2, int* __restrict__ idx, float* __restrict__ nxyz,
                             int S, int Np) {
  int t = blockIdx.x * blockDim.x + threadIdx.x;
  if (t >= Bc * S) return;
  int b = t / S;
  const float* P = xyz + (size_t)b * Np * 3;
  int f = fidx[t];
  float cx = P[f * 3 + 0], cy = P[f * 3 + 1], cz = P[f * 3 + 2];
  nxyz[t * 3 + 0] = cx; nxyz[t * 3 + 1] = cy; nxyz[t * 3 + 2] = cz;
  int buf[32]; int cnt = 0;
  for (int n = 0; n < Np && cnt < 32; ++n) {
    float dx = P[n * 3 + 0] - cx, dy = P[n * 3 + 1] - cy, dz = P[n * 3 + 2] - cz;
    if (dx * dx + dy * dy + dz * dz <= r2) buf[cnt++] = n;
  }
  int f0 = cnt > 0 ? buf[0] : 0;
  for (int k = cnt; k < 32; ++k) buf[k] = f0;
  for (int k = 0; k < 32; ++k) idx[t * 32 + k] = buf[k];
}

// ---------------------------------------------------------------- grouping -> f16 X with padded K stride
__global__ void k_group(const float* __restrict__ xyz, const float* __restrict__ nxyz,
                        const int* __restrict__ idx, const float* __restrict__ pts,
                        _Float16* __restrict__ X, int S, int Np, int Cp, int Kp) {
  int t = blockIdx.x * blockDim.x + threadIdx.x;
  if (t >= Bc * S * 32) return;
  int bs = t / 32; int b = bs / S;
  int n = idx[t];
  const float* P = xyz + ((size_t)b * Np + n) * 3;
  _Float16* x = X + (size_t)t * Kp;
  x[0] = (_Float16)(P[0] - nxyz[bs * 3 + 0]);
  x[1] = (_Float16)(P[1] - nxyz[bs * 3 + 1]);
  x[2] = (_Float16)(P[2] - nxyz[bs * 3 + 2]);
  const float* q = pts + ((size_t)b * Np + n) * Cp;
  for (int c = 0; c < Cp; ++c) x[3 + c] = (_Float16)q[c];
  for (int c = 3 + Cp; c < Kp; ++c) x[c] = (_Float16)0.f;
}

// ---------------------------------------------------------------- weight convert+pad f32(N,K) -> f16(N,Kp)
__global__ void k_cvt_w(const float* __restrict__ W, _Float16* __restrict__ Wh,
                        int Cout, int K, int Kp) {
  int t = blockIdx.x * blockDim.x + threadIdx.x;
  if (t >= Cout * Kp) return;
  int k = t % Kp; int o = t / Kp;
  Wh[t] = (k < K) ? (_Float16)W[(size_t)o * K + k] : (_Float16)0.f;
}

// ---------------------------------------------------------------- WMMA GEMM
// Y[M,Nn] = Xh[M,Kp](f16) * Wh[Nn,Kp]^T(f16) + bias.  M%16==0, Nn%(NSUB*16)==0,
// Kp%32==0.  One wave owns a 16 x (NSUB*16) supertile (A reused NSUB times).
// Branch-free inner loop: clause of global_load_b128 -> v_wmma per K-step.
template <int NSUB>
__global__ __launch_bounds__(256) void k_gemm(const _Float16* __restrict__ Xh,
                                              const _Float16* __restrict__ Wh,
                                              const float* __restrict__ bias,
                                              float* __restrict__ Y,
                                              int M, int Nn, int Kp) {
  int lane = threadIdx.x & 31;
  int wave = threadIdx.x >> 5;
  int tile = blockIdx.x * (blockDim.x >> 5) + wave;
  int tilesN = Nn / (NSUB * 16);
  int tilesM = M >> 4;
  if (tile >= tilesM * tilesN) return;
  int m0 = (tile / tilesN) << 4;
  int n0 = (tile % tilesN) * (NSUB * 16);
  int h = lane >> 4;
  int r = lane & 15;
  const _Float16* ap = Xh + (size_t)(m0 + r) * Kp;
  const _Float16* bp = Wh + (size_t)(n0 + r) * Kp + h * 16;
  size_t bstep = (size_t)16 * Kp;
  v8f acc[NSUB];
#pragma unroll
  for (int j = 0; j < NSUB; ++j) acc[j] = {};
  for (int k0 = 0; k0 < Kp; k0 += 32) {
    // A 16x32 f16 fragment: elem i -> K=k0+16h+i
    v16h a = *(const v16h*)(ap + k0 + h * 16);
    // B 32x16 f16 fragment: lane half h holds 16 consecutive K at column n+r
#pragma unroll
    for (int j = 0; j < NSUB; ++j) {
      v16h b = *(const v16h*)(bp + k0 + (size_t)j * bstep);
      acc[j] = __builtin_amdgcn_wmma_f32_16x16x32_f16(false, a, false, b, (short)0,
                                                      acc[j], false, false);
    }
  }
  // C/D layout: vgpr i -> row m0 + i + 8h, col n0 + 16j + (lane&15)
#pragma unroll
  for (int i = 0; i < 8; ++i) {
    int row = m0 + i + 8 * h;
    float* yr = Y + (size_t)row * Nn;
#pragma unroll
    for (int j = 0; j < NSUB; ++j)
      yr[n0 + 16 * j + r] = acc[j][i] + bias[n0 + 16 * j + r];
  }
}

// ---------------------------------------------------------------- batch-stat BN (+ReLU, f32 + f16 copy)
__global__ __launch_bounds__(256) void k_bn_stats(const float* __restrict__ Y,
                                                  float* __restrict__ mean,
                                                  float* __restrict__ var, int M, int Nn) {
  int n = blockIdx.x;
  __shared__ float ss[256], sq[256];
  float s = 0.f, q = 0.f;
  for (int m = threadIdx.x; m < M; m += blockDim.x) {
    float v = Y[(size_t)m * Nn + n];
    s += v; q += v * v;
  }
  ss[threadIdx.x] = s; sq[threadIdx.x] = q;
  __syncthreads();
  for (int off = 128; off > 0; off >>= 1) {
    if ((int)threadIdx.x < off) {
      ss[threadIdx.x] += ss[threadIdx.x + off];
      sq[threadIdx.x] += sq[threadIdx.x + off];
    }
    __syncthreads();
  }
  if (threadIdx.x == 0) {
    float mu = ss[0] / (float)M;
    mean[n] = mu;
    var[n]  = sq[0] / (float)M - mu * mu;
  }
}

__global__ void k_bn_relu(float* __restrict__ Y, _Float16* __restrict__ Yh,
                          const float* __restrict__ mean, const float* __restrict__ var,
                          const float* __restrict__ g, const float* __restrict__ be,
                          int M, int Nn) {
  int i = blockIdx.x * blockDim.x + threadIdx.x;
  if (i >= M * Nn) return;
  int n = i % Nn;
  float v = g[n] * (Y[i] - mean[n]) * rsqrtf(var[n] + 1e-5f) + be[n];
  v = v > 0.f ? v : 0.f;
  Y[i]  = v;
  Yh[i] = (_Float16)v;
}

// ---------------------------------------------------------------- max over 32 neighbors
__global__ void k_maxpool(const float* __restrict__ Xin, float* __restrict__ out,
                          int S, int C, int outStride, int outOff) {
  int t = blockIdx.x * blockDim.x + threadIdx.x;
  if (t >= Bc * S * C) return;
  int c = t % C; int bs = t / C;
  float m = -3.4e38f;
  for (int k = 0; k < 32; ++k) m = fmaxf(m, Xin[((size_t)bs * 32 + k) * C + c]);
  out[(size_t)bs * outStride + outOff + c] = m;
}

// ---------------------------------------------------------------- fp_layer: 3-NN interp + concat -> f16 X
__global__ void k_fp_interp(const float* __restrict__ xyz1, const float* __restrict__ xyz2,
                            const float* __restrict__ pts1, const float* __restrict__ pts2,
                            _Float16* __restrict__ X, int N1, int S2, int C1, int C2,
                            int Kp) {
  int t = blockIdx.x * blockDim.x + threadIdx.x;
  if (t >= Bc * N1) return;
  int b = t / N1;
  const float* p1 = xyz1 + (size_t)t * 3;
  float px = p1[0], py = p1[1], pz = p1[2];
  float d0 = 3.4e38f, d1 = 3.4e38f, d2 = 3.4e38f;
  int   i0 = 0, i1 = 0, i2 = 0;
  const float* Q = xyz2 + (size_t)b * S2 * 3;
  for (int j = 0; j < S2; ++j) {
    float dx = Q[j * 3 + 0] - px, dy = Q[j * 3 + 1] - py, dz = Q[j * 3 + 2] - pz;
    float d = dx * dx + dy * dy + dz * dz;
    if (d < d0)      { d2 = d1; i2 = i1; d1 = d0; i1 = i0; d0 = d; i0 = j; }
    else if (d < d1) { d2 = d1; i2 = i1; d1 = d;  i1 = j; }
    else if (d < d2) { d2 = d;  i2 = j; }
  }
  float w0 = 1.f / (fmaxf(d0, 0.f) + 1e-8f);
  float w1 = 1.f / (fmaxf(d1, 0.f) + 1e-8f);
  float w2 = 1.f / (fmaxf(d2, 0.f) + 1e-8f);
  float ws = w0 + w1 + w2; w0 /= ws; w1 /= ws; w2 /= ws;
  _Float16* x = X + (size_t)t * Kp;
  const float* q1 = pts1 + (size_t)t * C1;
  for (int c = 0; c < C1; ++c) x[c] = (_Float16)q1[c];
  const float* QB = pts2 + (size_t)b * S2 * C2;
  for (int c = 0; c < C2; ++c)
    x[C1 + c] = (_Float16)(w0 * QB[(size_t)i0 * C2 + c] + w1 * QB[(size_t)i1 * C2 + c] +
                           w2 * QB[(size_t)i2 * C2 + c]);
  for (int c = C1 + C2; c < Kp; ++c) x[c] = (_Float16)0.f;
}

// ---------------------------------------------------------------- final 128 -> 21 linear
__global__ void k_conv_out(const float* __restrict__ Xh, const float* __restrict__ W2,
                           const float* __restrict__ b2, float* __restrict__ out) {
  int t = blockIdx.x * blockDim.x + threadIdx.x;
  if (t >= Bc * NPts * NCls) return;
  int o = t % NCls; int bn = t / NCls;
  const float* x = Xh + (size_t)bn * 128;
  const float* w = W2 + (size_t)o * 128;
  float acc = b2[o];
  for (int c = 0; c < 128; ++c) acc += x[c] * w[c];
  out[t] = acc;
}

// ================================================================= host orchestration
struct MlpL { const float* W; const float* b; const float* g; const float* be; int c; };
struct FinP { float* y; _Float16* h; };

extern "C" void kernel_launch(void* const* d_in, const int* in_sizes, int n_in,
                              void* d_out, int out_size, void* d_ws, size_t ws_size,
                              hipStream_t stream) {
  (void)in_sizes; (void)n_in; (void)out_size; (void)ws_size;
  const float* xyz_in  = (const float*)d_in[0];   // (B,3,N)
  const float* pts_in  = (const float*)d_in[1];   // (B,3,N)
  const float* imageft = (const float*)d_in[2];   // (B,V,Cimg,H,W)
  const int*   idx3d   = (const int*)d_in[3];
  const int*   idx2d   = (const int*)d_in[4];

  int pi = 5;
  auto nextL = [&](int c) {
    MlpL L;
    L.W  = (const float*)d_in[pi + 0];
    L.b  = (const float*)d_in[pi + 1];
    L.g  = (const float*)d_in[pi + 2];
    L.be = (const float*)d_in[pi + 3];
    L.c  = c;
    pi += 4;
    return L;
  };
  MlpL sa1g[3] = { nextL(32),  nextL(32),  nextL(64)  };
  MlpL sa2g[3] = { nextL(64),  nextL(64),  nextL(128) };
  MlpL sa1f[3] = { nextL(32),  nextL(32),  nextL(64)  };
  MlpL sa2f[3] = { nextL(64),  nextL(64),  nextL(128) };
  MlpL sa3m[3] = { nextL(128), nextL(128), nextL(256) };
  MlpL sa4m[3] = { nextL(256), nextL(256), nextL(512) };
  MlpL fp4m[2] = { nextL(256), nextL(256) };
  MlpL fp3m[2] = { nextL(256), nextL(256) };
  MlpL fp2m[2] = { nextL(256), nextL(128) };
  MlpL fp1m[3] = { nextL(128), nextL(128), nextL(128) };
  MlpL headm[1] = { nextL(128) };
  const float* conv2w = (const float*)d_in[pi + 0];
  const float* conv2b = (const float*)d_in[pi + 1];

  // ---- workspace bump allocator (~85 MB peak) ----
  char* wsp = (char*)d_ws;
  auto A = [&](size_t nbytes) -> char* {
    char* p = wsp; wsp += (nbytes + 255) & ~(size_t)255; return p;
  };
  float* xyz_t = (float*)A(sizeof(float) * Bc * NPts * 3);
  float* pts_t = (float*)A(sizeof(float) * Bc * NPts * 3);
  float* imgf  = (float*)A(sizeof(float) * (size_t)Bc * NPts * Cimg);
  int*   jmax  = (int*)  A(sizeof(int) * Bc * Vv * NPts);
  float* l1x = (float*)A(sizeof(float) * Bc * 1024 * 3);
  float* l1p = (float*)A(sizeof(float) * Bc * 1024 * 64);
  float* f1x = (float*)A(sizeof(float) * Bc * 1024 * 3);
  float* f1p = (float*)A(sizeof(float) * Bc * 1024 * 64);
  float* l2x = (float*)A(sizeof(float) * Bc * 256 * 3);
  float* l2p = (float*)A(sizeof(float) * Bc * 256 * 256);
  float* f2x = (float*)A(sizeof(float) * Bc * 256 * 3);
  float* l3x = (float*)A(sizeof(float) * Bc * 64 * 3);
  float* l3p = (float*)A(sizeof(float) * Bc * 64 * 256);
  float* l4x = (float*)A(sizeof(float) * Bc * 16 * 3);
  float* l4p = (float*)A(sizeof(float) * Bc * 16 * 512);
  float* meanB = (float*)A(sizeof(float) * 512);
  float* varB  = (float*)A(sizeof(float) * 512);
  int*   fidxB = (int*)A(sizeof(int) * Bc * 1024);
  int*   idxB  = (int*)A(sizeof(int) * Bc * 1024 * 32);
  _Float16* Xbuf = (_Float16*)A(sizeof(_Float16) * (size_t)65536 * 160); // max rows*Kp
  _Float16* WhB  = (_Float16*)A(sizeof(_Float16) * (size_t)256 * 768);   // max Cout*Kp
  float*    Y0   = (float*)A(sizeof(float) * (size_t)65536 * 64);
  float*    Y1   = (float*)A(sizeof(float) * (size_t)65536 * 64);
  _Float16* H0   = (_Float16*)A(sizeof(_Float16) * (size_t)65536 * 64);
  _Float16* H1   = (_Float16*)A(sizeof(_Float16) * (size_t)65536 * 64);

  auto blk = [](int n) { return (n + 255) / 256; };

  // ---- building blocks ----
  auto gemm = [&](const _Float16* Xh, const MlpL& L, int K, int Kp, float* Y, int M) {
    k_cvt_w<<<blk(L.c * Kp), 256, 0, stream>>>(L.W, WhB, L.c, K, Kp);
    if (L.c % 64 == 0) {
      int tiles = (M / 16) * (L.c / 64);
      k_gemm<4><<<(tiles + 7) / 8, 256, 0, stream>>>(Xh, WhB, L.b, Y, M, L.c, Kp);
    } else {            // Nn == 32
      int tiles = (M / 16) * (L.c / 32);
      k_gemm<2><<<(tiles + 7) / 8, 256, 0, stream>>>(Xh, WhB, L.b, Y, M, L.c, Kp);
    }
  };
  auto mlp_chain = [&](const _Float16* Xin, int M, int K0, int Kp0, const MlpL* L, int nl,
                       float* Ya, float* Yb, _Float16* Ha, _Float16* Hb) -> FinP {
    const _Float16* cur = Xin; int K = K0, Kp = Kp0;
    FinP fin = { Ya, Ha };
    for (int i = 0; i < nl; ++i) {
      float* Y = (i & 1) ? Yb : Ya;
      _Float16* H = (i & 1) ? Hb : Ha;
      gemm(cur, L[i], K, Kp, Y, M);
      k_bn_stats<<<L[i].c, 256, 0, stream>>>(Y, meanB, varB, M, L[i].c);
      k_bn_relu<<<blk(M * L[i].c), 256, 0, stream>>>(Y, H, meanB, varB, L[i].g, L[i].be,
                                                     M, L[i].c);
      cur = H; K = Kp = L[i].c; fin.y = Y; fin.h = H;
    }
    return fin;
  };
  auto sa = [&](const float* xyz, const float* pts, int Np, int S, float radius, int Cp,
                const MlpL* L, int nl, float* outXyz, float* outP, int outStride, int outOff) {
    k_fps<<<Bc, 256, 0, stream>>>(xyz, fidxB, Np, S);
    k_query_ball<<<blk(Bc * S), 256, 0, stream>>>(xyz, fidxB, radius * radius, idxB,
                                                  outXyz, S, Np);
    int rows = Bc * S * 32;
    int Kp = padK(3 + Cp);
    k_group<<<blk(rows), 256, 0, stream>>>(xyz, outXyz, idxB, pts, Xbuf, S, Np, Cp, Kp);
    FinP fin = mlp_chain(Xbuf, rows, 3 + Cp, Kp, L, nl, Y0, Y1, H0, H1);
    int C = L[nl - 1].c;
    k_maxpool<<<blk(Bc * S * C), 256, 0, stream>>>(fin.y, outP, S, C, outStride, outOff);
  };
  auto fp = [&](const float* xyz1, const float* xyz2, const float* p1, const float* p2,
                int N1, int S2, int C1, int C2, const MlpL* L, int nl) -> FinP {
    int Kp = padK(C1 + C2);
    k_fp_interp<<<blk(Bc * N1), 256, 0, stream>>>(xyz1, xyz2, p1, p2, Xbuf, N1, S2,
                                                  C1, C2, Kp);
    return mlp_chain(Xbuf, Bc * N1, C1 + C2, Kp, L, nl, Y0, Y1, H0, H1);
  };

  // ---- stage 0: transposes + project_views ----
  k_transpose<<<blk(Bc * 3 * NPts), 256, 0, stream>>>(xyz_in, xyz_t, 3, NPts);
  k_transpose<<<blk(Bc * 3 * NPts), 256, 0, stream>>>(pts_in, pts_t, 3, NPts);
  k_fill_i32<<<blk(Bc * Vv * NPts), 256, 0, stream>>>(jmax, -1, Bc * Vv * NPts);
  k_scatter_last<<<blk(Bc * Vv * NPts), 256, 0, stream>>>(idx3d, jmax);
  k_project<<<blk(Bc * NPts * Cimg), 256, 0, stream>>>(imageft, idx2d, jmax, imgf);

  // ---- set-abstraction pyramid ----
  sa(xyz_t, pts_t, NPts, 1024, 0.1f, 3,   sa1g, 3, l1x, l1p, 64,  0);
  sa(l1x,   l1p,   1024, 256,  0.2f, 64,  sa2g, 3, l2x, l2p, 256, 0);
  sa(xyz_t, imgf,  NPts, 1024, 0.1f, 128, sa1f, 3, f1x, f1p, 64,  0);
  sa(f1x,   f1p,   1024, 256,  0.2f, 64,  sa2f, 3, f2x, l2p, 256, 128); // concat -> l2p[:,128:]
  sa(l2x,   l2p,   256,  64,   0.4f, 256, sa3m, 3, l3x, l3p, 256, 0);
  sa(l3x,   l3p,   64,   16,   0.8f, 256, sa4m, 3, l4x, l4p, 512, 0);

  // ---- feature propagation (fin ping-pongs through Y0/Y1 + H0/H1; stream order keeps it safe) ----
  FinP p3 = fp(l3x, l4x, l3p, l4p, 64,   16,  256, 512, fp4m, 2);   // -> Y1/H1 (B,64,256)
  FinP p2 = fp(l2x, l3x, l2p, p3.y, 256, 64,  256, 256, fp3m, 2);   // -> Y1/H1 (B,256,256)
  FinP p1 = fp(l1x, l2x, l1p, p2.y, 1024, 256, 64, 256, fp2m, 2);   // -> Y1/H1 (B,1024,128)
  FinP p0 = fp(xyz_t, l1x, pts_t, p1.y, NPts, 1024, 3, 128, fp1m, 3); // -> Y0/H0 (B,8192,128)

  // ---- head (input H0, so chain through Y1/H1) + classifier ----
  FinP ph = mlp_chain(p0.h, Bc * NPts, 128, 128, headm, 1, Y1, Y0, H1, H0); // -> Y1
  k_conv_out<<<blk(Bc * NPts * NCls), 256, 0, stream>>>(ph.y, conv2w, conv2b,
                                                        (float*)d_out);
}
